// SageFormer_23837068492824
// MI455X (gfx1250) — matrix operand order repaired and checked by reading, not compiled, not run on hardware
//
#include <hip/hip_runtime.h>
#include <hip/hip_bf16.h>
#include <math.h>

typedef __attribute__((ext_vector_type(16))) _Float16 v16h;
typedef __attribute__((ext_vector_type(8)))  _Float16 v8h;
typedef __attribute__((ext_vector_type(8)))  float    v8f;
typedef __attribute__((ext_vector_type(4))) unsigned int u32x4;
typedef __attribute__((ext_vector_type(4))) int i32x4;
typedef __attribute__((ext_vector_type(8))) int i32x8;

#define BSZ   64
#define TLEN  96
#define CCH   16
#define KIN   16
#define HDIM  256
#define NHEAD 8
#define DHEAD 32
#define NCLS  4
#define LSEQ  100     // TLEN + NCLS
#define MROWS (1024 * LSEQ)   // (BSZ*CCH) * LSEQ = 102400

#if defined(__has_builtin)
#  if __has_builtin(__builtin_amdgcn_tensor_load_to_lds) && __has_builtin(__builtin_amdgcn_s_wait_tensorcnt)
#    define HAVE_TDM 1
#  endif
#endif

// ---------------------------------------------------------------- helpers
__device__ __forceinline__ v16h pack16(v8h lo, v8h hi) {
  v16h r;
#pragma unroll
  for (int i = 0; i < 8; ++i) { r[i] = lo[i]; r[i + 8] = hi[i]; }
  return r;
}
__device__ __forceinline__ v16h zero16() {
  v16h r;
#pragma unroll
  for (int i = 0; i < 16; ++i) r[i] = (_Float16)0.f;
  return r;
}
__device__ __forceinline__ v8f zero8f() {
  v8f r;
#pragma unroll
  for (int i = 0; i < 8; ++i) r[i] = 0.f;
  return r;
}

#ifdef HAVE_TDM
// TDM 2D tile load: rows x rowlen_h (halfs) tile, global row stride gstride_h (halfs),
// packed contiguously into LDS. D# bitfields per CDNA5 ISA 8.3-8.5 (data_size=2B, type=2).
__device__ __forceinline__ void tdm_load_2d(const _Float16* gsrc, void* lds_dst,
                                            int rows, int rowlen_h, int gstride_h) {
  unsigned long long ga = (unsigned long long)gsrc;
  unsigned lds = (unsigned)(unsigned long long)lds_dst;   // LDS byte offset = addr[31:0]
  u32x4 g0;
  g0[0] = 1u;                                       // count=1 user descriptor
  g0[1] = lds;                                      // lds_addr (bytes)
  g0[2] = (unsigned)(ga & 0xffffffffu);             // global_addr[31:0]
  g0[3] = (unsigned)((ga >> 32) & 0x01ffffffu)      // global_addr[56:32]
        | (2u << 30);                               // type=2 ("image")
  i32x8 g1;
  g1[0] = (1 << 16);                                // data_size=1 -> 2 bytes
  g1[1] = (rowlen_h & 0xffff) << 16;                // tensor_dim0[15:0]
  g1[2] = ((unsigned)rowlen_h >> 16)                // tensor_dim0[31:16]
        | ((rows & 0xffff) << 16);                  // tensor_dim1[15:0]
  g1[3] = ((unsigned)rows >> 16)                    // tensor_dim1[31:16]
        | ((rowlen_h & 0xffff) << 16);              // tile_dim0
  g1[4] = (rows & 0xffff);                          // tile_dim1 (tile_dim2 = 0)
  g1[5] = gstride_h;                                // tensor_dim0_stride[31:0]
  g1[6] = 0;                                        // stride0[47:32] | stride1[15:0]
  g1[7] = 0;
  i32x4 z4;
  z4[0] = z4[1] = z4[2] = z4[3] = 0;
#if __clang_major__ >= 23
  i32x8 z8;
#pragma unroll
  for (int i = 0; i < 8; ++i) z8[i] = 0;
  __builtin_amdgcn_tensor_load_to_lds(g0, g1, z4, z4, z8, 0);
#else
  __builtin_amdgcn_tensor_load_to_lds(g0, g1, z4, z4, 0);
#endif
}
#endif

// ---------------------------------------------------------------- fp32 -> fp16
__global__ void k_cvt(const float* __restrict__ s, _Float16* __restrict__ d, int n) {
  int i = blockIdx.x * blockDim.x + threadIdx.x;
  if (i < n) d[i] = (_Float16)s[i];
}

// ---------------------------------------------------------------- graph constructor (tiny)
__global__ void k_graph(const float* __restrict__ emb1, const float* __restrict__ emb2,
                        const float* __restrict__ Wl1, const float* __restrict__ bl1,
                        const float* __restrict__ Wl2, const float* __restrict__ bl2,
                        float* __restrict__ aout /*16x16 normalized adjacency*/) {
  __shared__ float m1[CCH][HDIM];
  __shared__ float m2[CCH][HDIM];
  __shared__ float adj[CCH][CCH];
  __shared__ float rs[CCH];
  const int t = threadIdx.x;
  for (int idx = t; idx < CCH * HDIM; idx += blockDim.x) {
    int c = idx >> 8, h = idx & 255;
    float s1 = bl1[h], s2 = bl2[h];
    for (int k = 0; k < HDIM; ++k) {
      s1 += emb1[c * HDIM + k] * Wl1[h * HDIM + k];
      s2 += emb2[c * HDIM + k] * Wl2[h * HDIM + k];
    }
    m1[c][h] = tanhf(s1);
    m2[c][h] = tanhf(s2);
  }
  __syncthreads();
  if (t < 256) {
    int v = t >> 4, w = t & 15;
    float s = 0.f;
    for (int k = 0; k < HDIM; ++k) s += m1[v][k] * m2[w][k] - m2[v][k] * m1[w][k];
    float g = fmaxf(tanhf(s), 0.f);
    if (v == w) g += 1.f;
    adj[v][w] = g;
  }
  __syncthreads();
  if (t < CCH) {
    float s = 0.f;
    for (int w = 0; w < CCH; ++w) s += adj[t][w];
    rs[t] = 1.f / s;
  }
  __syncthreads();
  if (t < 256) aout[t] = adj[t >> 4][t & 15] * rs[t >> 4];
}

// ---------------------------------------------------------------- embed: x@Wfc^T + bfc | cls, + pos
__global__ void k_embed(const float* __restrict__ x, const float* __restrict__ Wfc,
                        const float* __restrict__ bfc, const float* __restrict__ cls,
                        const float* __restrict__ pos,
                        float* __restrict__ h32, _Float16* __restrict__ h16) {
  const int row = blockIdx.x;              // bc*LSEQ + l
  const int bc  = row / LSEQ, l = row % LSEQ;
  const int b   = bc >> 4, c = bc & 15;
  const int hd  = threadIdx.x;
  float v;
  if (l < TLEN) {
    const float* xr = x + (((size_t)b * TLEN + l) * CCH + c) * KIN;
    const float* wr = Wfc + (size_t)hd * KIN;
    float s = bfc[hd];
#pragma unroll
    for (int i = 0; i < KIN; ++i) s += xr[i] * wr[i];
    v = s;
  } else {
    v = cls[(((size_t)(l - TLEN)) * CCH + c) * HDIM + hd];
  }
  v += pos[((size_t)l * CCH + c) * HDIM + hd];
  h32[(size_t)row * HDIM + hd] = v;
  h16[(size_t)row * HDIM + hd] = (_Float16)v;
}

// ---------------------------------------------------------------- WMMA GEMM: out = act(A(MxK) * B(NxK)^T + bias [+ res])
// block = 256 threads (8 waves); block tile 128(M) x 64(N); wave tile 16 x 64.
// Two-stage ping-pong pipeline over K (K must be a multiple of 64): each stage's
// fragment loads overlap the other stage's WMMAs, with no register rotation copies.
__global__ void __launch_bounds__(256)
k_gemm(const _Float16* __restrict__ A, const _Float16* __restrict__ B,
       const float* __restrict__ bias, const float* __restrict__ res,
       float* __restrict__ outF, _Float16* __restrict__ outH,
       int M, int N, int K, int act /*0 none, 1 relu, 2 gelu*/) {
  const int lane = threadIdx.x & 31;
  const int wv   = threadIdx.x >> 5;
  const int lh   = lane >> 4;        // lane half (0: lanes 0-15, 1: lanes 16-31)
  const int l16  = lane & 15;
  const int mBase = blockIdx.y * 128 + wv * 16;
  const int N0    = blockIdx.x * 64;
  const int mRow  = mBase + l16;
  const _Float16* arow = A + (size_t)(mRow < M ? mRow : 0) * K;
  const int aoff0 = lh ? 8 : 0;
  const int aoff1 = lh ? 24 : 16;
  const int boff  = lh ? 8 : 0;

  v8f acc[4];
#pragma unroll
  for (int t = 0; t < 4; ++t) acc[t] = zero8f();

  auto loadA = [&](int k0) -> v16h {
    v8h lo = *(const v8h*)(arow + k0 + aoff0);
    v8h hi = *(const v8h*)(arow + k0 + aoff1);
    return pack16(lo, hi);
  };
  auto loadB = [&](int t, int k0) -> v16h {
    const _Float16* brow = B + (size_t)(N0 + t * 16 + l16) * K + k0 + boff;
    v8h lo = *(const v8h*)brow;
    v8h hi = *(const v8h*)(brow + 16);
    return pack16(lo, hi);
  };
#define WMMA_F16(a, b, c) __builtin_amdgcn_wmma_f32_16x16x32_f16(false, a, false, b, (short)0, c, false, false)

  // stage 0 fragments for k = 0
  v16h a0  = loadA(0);
  v16h b00 = loadB(0, 0), b01 = loadB(1, 0), b02 = loadB(2, 0), b03 = loadB(3, 0);

  int k0 = 0;
  for (; k0 + 64 < K; k0 += 64) {
    __builtin_prefetch(arow + k0 + 64, 0, 1);            // global_prefetch_b8
    // stage 1 loads (k0+32) overlap stage-0 WMMAs (k0)
    v16h a1  = loadA(k0 + 32);
    v16h b10 = loadB(0, k0 + 32), b11 = loadB(1, k0 + 32),
         b12 = loadB(2, k0 + 32), b13 = loadB(3, k0 + 32);
    acc[0] = WMMA_F16(a0, b00, acc[0]);
    acc[1] = WMMA_F16(a0, b01, acc[1]);
    acc[2] = WMMA_F16(a0, b02, acc[2]);
    acc[3] = WMMA_F16(a0, b03, acc[3]);
    // stage 0 loads (k0+64) overlap stage-1 WMMAs (k0+32)
    a0  = loadA(k0 + 64);
    b00 = loadB(0, k0 + 64); b01 = loadB(1, k0 + 64);
    b02 = loadB(2, k0 + 64); b03 = loadB(3, k0 + 64);
    acc[0] = WMMA_F16(a1, b10, acc[0]);
    acc[1] = WMMA_F16(a1, b11, acc[1]);
    acc[2] = WMMA_F16(a1, b12, acc[2]);
    acc[3] = WMMA_F16(a1, b13, acc[3]);
  }
  // tail: k0 == K - 64; stage 0 already holds (K-64) fragments
  {
    v16h a1  = loadA(k0 + 32);
    v16h b10 = loadB(0, k0 + 32), b11 = loadB(1, k0 + 32),
         b12 = loadB(2, k0 + 32), b13 = loadB(3, k0 + 32);
    acc[0] = WMMA_F16(a0, b00, acc[0]);
    acc[1] = WMMA_F16(a0, b01, acc[1]);
    acc[2] = WMMA_F16(a0, b02, acc[2]);
    acc[3] = WMMA_F16(a0, b03, acc[3]);
    acc[0] = WMMA_F16(a1, b10, acc[0]);
    acc[1] = WMMA_F16(a1, b11, acc[1]);
    acc[2] = WMMA_F16(a1, b12, acc[2]);
    acc[3] = WMMA_F16(a1, b13, acc[3]);
  }
#undef WMMA_F16

#pragma unroll
  for (int t = 0; t < 4; ++t) {
    const int n = N0 + t * 16 + l16;
    const float bn = bias ? bias[n] : 0.f;
#pragma unroll
    for (int i = 0; i < 8; ++i) {
      const int m = mBase + lh * 8 + i;
      if (m < M && n < N) {
        float v = acc[t][i] + bn;
        if (res) v += res[(size_t)m * N + n];
        if (act == 1)      v = fmaxf(v, 0.f);
        else if (act == 2) v = 0.5f * v * (1.f + erff(v * 0.70710678118f));
        if (outH) outH[(size_t)m * N + n] = (_Float16)v;
        else      outF[(size_t)m * N + n] = v;
      }
    }
  }
}

// ---------------------------------------------------------------- attention: one block per (seq, head)
// 7 waves; wave w handles query rows [16w,16w+16). L padded to 112 (scores) / 128 (PV K-dim).
__global__ void k_attn(const _Float16* __restrict__ qkv, _Float16* __restrict__ o16) {
  __shared__ __align__(16) _Float16 Kl[112][32];
  __shared__ __align__(16) _Float16 Vl[128][32];
  __shared__ __align__(16) _Float16 Pst[7][16][128];

  const int bc  = blockIdx.x >> 3;   // 0..1023
  const int hd  = blockIdx.x & 7;
  const int tid = threadIdx.x;       // 224 threads
  const int wv  = tid >> 5;
  const int lane = tid & 31;
  const int lh  = lane >> 4;
  const int l16 = lane & 15;

#ifdef HAVE_TDM
  // Tensor Data Mover: DMA the 100x32-half K and V tiles (row stride 768 halfs)
  // straight into LDS; one issue per TDM op from wave 0 (TDM ignores EXEC).
  if (wv == 0) {
    const _Float16* base = qkv + ((size_t)bc * LSEQ) * 768 + hd * DHEAD;
    tdm_load_2d(base + 256, &Kl[0][0], LSEQ, DHEAD, 768);
    tdm_load_2d(base + 512, &Vl[0][0], LSEQ, DHEAD, 768);
    __builtin_amdgcn_s_wait_tensorcnt(0);
  }
  // zero the padding rows (TDM fills rows 0..99 only)
  for (int idx = tid; idx < 12 * 32; idx += blockDim.x)
    Kl[100 + (idx >> 5)][idx & 31] = (_Float16)0.f;
  for (int idx = tid; idx < 28 * 32; idx += blockDim.x)
    Vl[100 + (idx >> 5)][idx & 31] = (_Float16)0.f;
#else
  for (int idx = tid; idx < 128 * 32; idx += blockDim.x) {
    int l = idx >> 5, d = idx & 31;
    _Float16 kv = (_Float16)0.f, vv = (_Float16)0.f;
    if (l < LSEQ) {
      size_t rb = ((size_t)bc * LSEQ + l) * 768 + hd * DHEAD + d;
      kv = qkv[rb + 256];
      vv = qkv[rb + 512];
    }
    if (l < 112) Kl[l][d] = kv;
    Vl[l][d] = vv;
  }
#endif
  // zero the P-staging tail columns (112..127)
  for (int idx = tid; idx < 7 * 16 * 16; idx += blockDim.x) {
    int w = idx >> 8, r = idx & 255;
    Pst[w][r >> 4][112 + (r & 15)] = (_Float16)0.f;
  }
  __syncthreads();

  const int qm = wv * 16 + l16;

  // Q fragment (A layout) straight from global
  v16h qa;
  if (qm < LSEQ) {
    const _Float16* qr = qkv + ((size_t)bc * LSEQ + qm) * 768 + hd * DHEAD;
    v8h lo = *(const v8h*)(qr + (lh ? 8 : 0));
    v8h hi = *(const v8h*)(qr + (lh ? 24 : 16));
    qa = pack16(lo, hi);
  } else {
    qa = zero16();
  }

  // scores: S[t] = Q * K_tile^T  (K dim = 32)
  v8f s[7];
#pragma unroll
  for (int t = 0; t < 7; ++t) {
    const int kn = t * 16 + l16;
    const _Float16* kr = &Kl[kn][lh * 8];
    v8h lo = *(const v8h*)kr;
    v8h hi = *(const v8h*)(kr + 16);
    v16h kf = pack16(lo, hi);
    s[t] = __builtin_amdgcn_wmma_f32_16x16x32_f16(false, qa, false, kf,
                                                  (short)0, zero8f(), false, false);
  }

  // scale + mask + softmax (row = m, spread across 16 lanes per half-group)
  const float scale = 0.17677669529663687f;  // 1/sqrt(32)
  float mx[8];
#pragma unroll
  for (int i = 0; i < 8; ++i) mx[i] = -1e30f;
#pragma unroll
  for (int t = 0; t < 7; ++t)
#pragma unroll
    for (int i = 0; i < 8; ++i) {
      const int lq = wv * 16 + lh * 8 + i;
      const int lk = t * 16 + l16;
      float v = s[t][i] * scale;
      const bool ok = (lk < LSEQ) && (lq >= TLEN || lk <= lq);
      v = ok ? v : -1e30f;
      s[t][i] = v;
      mx[i] = fmaxf(mx[i], v);
    }
#pragma unroll
  for (int i = 0; i < 8; ++i)
#pragma unroll
    for (int o = 8; o > 0; o >>= 1) mx[i] = fmaxf(mx[i], __shfl_xor(mx[i], o, 32));

  float sum[8];
#pragma unroll
  for (int i = 0; i < 8; ++i) sum[i] = 0.f;
#pragma unroll
  for (int t = 0; t < 7; ++t)
#pragma unroll
    for (int i = 0; i < 8; ++i) {
      float e = __expf(s[t][i] - mx[i]);
      s[t][i] = e;
      sum[i] += e;
    }
#pragma unroll
  for (int i = 0; i < 8; ++i) {
#pragma unroll
    for (int o = 8; o > 0; o >>= 1) sum[i] += __shfl_xor(sum[i], o, 32);
    sum[i] = 1.f / sum[i];
  }

  // write normalized P into per-wave LDS staging (plain row-major, transposes C->A layout)
#pragma unroll
  for (int t = 0; t < 7; ++t)
#pragma unroll
    for (int i = 0; i < 8; ++i)
      Pst[wv][lh * 8 + i][t * 16 + l16] = (_Float16)(s[t][i] * sum[i]);
  __syncthreads();

  // O = P(16x128) * V(128x32), K chunks of 32
  v8f oacc[2];
  oacc[0] = zero8f();
  oacc[1] = zero8f();
#pragma unroll
  for (int kc = 0; kc < 4; ++kc) {
    const _Float16* prow = &Pst[wv][l16][kc * 32];
    v8h lo = *(const v8h*)(prow + (lh ? 16 : 0));
    v8h hi = *(const v8h*)(prow + (lh ? 24 : 8));
    v16h pa = pack16(lo, hi);
    const int kb0 = kc * 32 + (lh ? 16 : 0);
#pragma unroll
    for (int nt = 0; nt < 2; ++nt) {
      v16h vb;
#pragma unroll
      for (int e = 0; e < 16; ++e) vb[e] = Vl[kb0 + e][nt * 16 + l16];
      oacc[nt] = __builtin_amdgcn_wmma_f32_16x16x32_f16(false, pa, false, vb,
                                                        (short)0, oacc[nt], false, false);
    }
  }
#pragma unroll
  for (int nt = 0; nt < 2; ++nt)
#pragma unroll
    for (int i = 0; i < 8; ++i) {
      const int lq = wv * 16 + lh * 8 + i;
      if (lq < LSEQ)
        o16[((size_t)bc * LSEQ + lq) * HDIM + hd * DHEAD + nt * 16 + l16] =
            (_Float16)oacc[nt][i];
    }
}

// ---------------------------------------------------------------- LayerNorm (one block per row, 256 threads)
__global__ void k_ln(const float* __restrict__ in, const float* __restrict__ g,
                     const float* __restrict__ b, float* __restrict__ o32,
                     _Float16* __restrict__ o16) {
  __shared__ float red[8];
  __shared__ float bc0;
  const int row = blockIdx.x;
  const int t = threadIdx.x;
  const float v = in[(size_t)row * HDIM + t];

  float s = v;
#pragma unroll
  for (int o = 16; o > 0; o >>= 1) s += __shfl_xor(s, o, 32);
  if ((t & 31) == 0) red[t >> 5] = s;
  __syncthreads();
  if (t == 0) { float a = 0; for (int i = 0; i < 8; ++i) a += red[i]; bc0 = a * (1.f / HDIM); }
  __syncthreads();
  const float mean = bc0;
  __syncthreads();

  const float d = v - mean;
  s = d * d;
#pragma unroll
  for (int o = 16; o > 0; o >>= 1) s += __shfl_xor(s, o, 32);
  if ((t & 31) == 0) red[t >> 5] = s;
  __syncthreads();
  if (t == 0) { float a = 0; for (int i = 0; i < 8; ++i) a += red[i]; bc0 = a * (1.f / HDIM); }
  __syncthreads();
  const float rstd = rsqrtf(bc0 + 1e-5f);

  const float out = d * rstd * g[t] + b[t];
  o32[(size_t)row * HDIM + t] = out;
  o16[(size_t)row * HDIM + t] = (_Float16)out;
}

// ---------------------------------------------------------------- mixprop on cls tokens (one block per (n,l))
__global__ void k_mixprop(float* __restrict__ h32, _Float16* __restrict__ h16,
                          const float* __restrict__ a /*16x16*/,
                          const float* __restrict__ Wm /*H x 3H*/,
                          const float* __restrict__ bm /*H*/) {
  __shared__ float x0[CCH][HDIM];
  __shared__ float x1[CCH][HDIM];
  __shared__ float x2[CCH][HDIM];
  __shared__ float as[256];
  const int n = blockIdx.x >> 2;
  const int l = blockIdx.x & 3;
  const int t = threadIdx.x;   // hdim
  as[t] = a[t];
  for (int c = 0; c < CCH; ++c)
    x0[c][t] = h32[(((size_t)(n * CCH + c)) * LSEQ + TLEN + l) * HDIM + t];
  __syncthreads();
  for (int v = 0; v < CCH; ++v) {
    float s = 0.f;
#pragma unroll
    for (int w = 0; w < CCH; ++w) s += x0[w][t] * as[v * 16 + w];
    x1[v][t] = s;
  }
  for (int v = 0; v < CCH; ++v) {
    float s = 0.f;
#pragma unroll
    for (int w = 0; w < CCH; ++w) s += x1[w][t] * as[v * 16 + w];
    x2[v][t] = s;
  }
  __syncthreads();
  float outv[CCH];
  const float* wr = Wm + (size_t)t * 3 * HDIM;
  for (int v = 0; v < CCH; ++v) {
    float sacc = bm[t];
    for (int h = 0; h < HDIM; ++h)
      sacc += wr[h] * x0[v][h] + wr[HDIM + h] * x1[v][h] + wr[2 * HDIM + h] * x2[v][h];
    outv[v] = sacc;
  }
  __syncthreads();
  for (int v = 0; v < CCH; ++v) {
    const size_t idx = (((size_t)(n * CCH + v)) * LSEQ + TLEN + l) * HDIM + t;
    h32[idx] = outv[v];
    h16[idx] = (_Float16)outv[v];
  }
}

// ---------------------------------------------------------------- head: tanh-pack then tiny GEMM + dot
__global__ void k_pack(const float* __restrict__ h32, _Float16* __restrict__ z16) {
  const int i = blockIdx.x * blockDim.x + threadIdx.x;
  if (i >= BSZ * 16384) return;
  const int n = i >> 14, j = i & 16383;
  const int c = j >> 10, r = j & 1023, lc = r >> 8, hd = r & 255;
  const float v = h32[(((size_t)(n * CCH + c)) * LSEQ + TLEN + lc) * HDIM + hd];
  z16[i] = (_Float16)tanhf(v);
}

__global__ void k_head(const float* __restrict__ d1, const float* __restrict__ Wd2,
                       const float* __restrict__ bd2, float* __restrict__ out) {
  __shared__ float red[8];
  const int n = blockIdx.x;
  const int t = threadIdx.x;
  float s = d1[(size_t)n * HDIM + t] * Wd2[t];
#pragma unroll
  for (int o = 16; o > 0; o >>= 1) s += __shfl_xor(s, o, 32);
  if ((t & 31) == 0) red[t >> 5] = s;
  __syncthreads();
  if (t == 0) {
    float a = 0;
    for (int i = 0; i < 8; ++i) a += red[i];
    out[n] = a + bd2[0];
  }
}

// ---------------------------------------------------------------- launcher
extern "C" void kernel_launch(void* const* d_in, const int* in_sizes, int n_in,
                              void* d_out, int out_size, void* d_ws, size_t ws_size,
                              hipStream_t stream) {
  (void)in_sizes; (void)n_in; (void)out_size; (void)ws_size;
  const float* x    = (const float*)d_in[0];
  const float* Wfc  = (const float*)d_in[3];
  const float* bfc  = (const float*)d_in[4];
  const float* cls  = (const float*)d_in[5];
  const float* pos  = (const float*)d_in[6];
  const float* emb1 = (const float*)d_in[7];
  const float* emb2 = (const float*)d_in[8];
  const float* Wl1  = (const float*)d_in[9];
  const float* bl1  = (const float*)d_in[10];
  const float* Wl2  = (const float*)d_in[11];
  const float* bl2  = (const float*)d_in[12];
  const float* Wqkv = (const float*)d_in[13];
  const float* bqkv = (const float*)d_in[14];
  const float* Wo   = (const float*)d_in[15];
  const float* bo   = (const float*)d_in[16];
  const float* W1   = (const float*)d_in[17];
  const float* b1   = (const float*)d_in[18];
  const float* W2   = (const float*)d_in[19];
  const float* b2   = (const float*)d_in[20];
  const float* ln1g = (const float*)d_in[21];
  const float* ln1b = (const float*)d_in[22];
  const float* ln2g = (const float*)d_in[23];
  const float* ln2b = (const float*)d_in[24];
  const float* Wmlp = (const float*)d_in[25];
  const float* bmlp = (const float*)d_in[26];
  const float* Wd1  = (const float*)d_in[27];
  const float* bd1  = (const float*)d_in[28];
  const float* Wd2  = (const float*)d_in[29];
  const float* bd2  = (const float*)d_in[30];

  char* ws = (char*)d_ws;
  size_t off = 0;
  auto alloc = [&](size_t bytes) -> char* {
    char* p = ws + off;
    off += (bytes + 255) & ~(size_t)255;
    return p;
  };
  const int M = MROWS;
  float*    h32   = (float*)alloc((size_t)M * HDIM * 4);
  _Float16* h16   = (_Float16*)alloc((size_t)M * HDIM * 2);
  _Float16* qkv16 = (_Float16*)alloc((size_t)M * 768 * 2);
  _Float16* sc16  = (_Float16*)alloc((size_t)M * 1024 * 2);   // attn-out then ffn-hidden
  float*    tmp32 = (float*)alloc((size_t)M * HDIM * 4);
  _Float16* z16   = (_Float16*)alloc((size_t)BSZ * 16384 * 2);
  float*    d1o   = (float*)alloc((size_t)BSZ * HDIM * 4);
  float*    aadj  = (float*)alloc(256 * 4);
  _Float16* wq16  = (_Float16*)alloc((size_t)4 * 768 * 256 * 2);
  _Float16* wo16  = (_Float16*)alloc((size_t)4 * 256 * 256 * 2);
  _Float16* w116  = (_Float16*)alloc((size_t)4 * 1024 * 256 * 2);
  _Float16* w216  = (_Float16*)alloc((size_t)4 * 256 * 1024 * 2);
  _Float16* wd116 = (_Float16*)alloc((size_t)256 * 16384 * 2);

  auto cvt = [&](const float* s, _Float16* d, int n) {
    k_cvt<<<(n + 255) / 256, 256, 0, stream>>>(s, d, n);
  };
  cvt(Wqkv, wq16, 4 * 768 * 256);
  cvt(Wo,   wo16, 4 * 256 * 256);
  cvt(W1,   w116, 4 * 1024 * 256);
  cvt(W2,   w216, 4 * 256 * 1024);
  cvt(Wd1,  wd116, 256 * 16384);

  k_graph<<<1, 256, 0, stream>>>(emb1, emb2, Wl1, bl1, Wl2, bl2, aadj);
  k_embed<<<M, 256, 0, stream>>>(x, Wfc, bfc, cls, pos, h32, h16);

  for (int l = 0; l < 4; ++l) {
    if (l > 0)
      k_mixprop<<<BSZ * NCLS, 256, 0, stream>>>(h32, h16, aadj,
          Wmlp + (size_t)(l - 1) * 256 * 768, bmlp + (size_t)(l - 1) * 256);
    // QKV projection -> fp16
    k_gemm<<<dim3(768 / 64, M / 128), 256, 0, stream>>>(
        h16, wq16 + (size_t)l * 768 * 256, bqkv + l * 768,
        nullptr, nullptr, qkv16, M, 768, 256, 0);
    // attention -> fp16
    k_attn<<<1024 * NHEAD, 224, 0, stream>>>(qkv16, sc16);
    // O-proj + residual -> fp32
    k_gemm<<<dim3(256 / 64, M / 128), 256, 0, stream>>>(
        sc16, wo16 + (size_t)l * 256 * 256, bo + l * 256,
        h32, tmp32, nullptr, M, 256, 256, 0);
    k_ln<<<M, 256, 0, stream>>>(tmp32, ln1g + l * 256, ln1b + l * 256, h32, h16);
    // FFN
    k_gemm<<<dim3(1024 / 64, M / 128), 256, 0, stream>>>(
        h16, w116 + (size_t)l * 1024 * 256, b1 + l * 1024,
        nullptr, nullptr, sc16, M, 1024, 256, 1);
    k_gemm<<<dim3(256 / 64, M / 128), 256, 0, stream>>>(
        sc16, w216 + (size_t)l * 256 * 1024, b2 + l * 256,
        h32, tmp32, nullptr, M, 256, 1024, 0);
    k_ln<<<M, 256, 0, stream>>>(tmp32, ln2g + l * 256, ln2b + l * 256, h32, h16);
  }

  k_pack<<<(BSZ * 16384) / 256, 256, 0, stream>>>(h32, z16);
  k_gemm<<<dim3(256 / 64, 1), 256, 0, stream>>>(
      z16, wd116, bd1, nullptr, d1o, nullptr, BSZ, 256, 16384, 2);
  k_head<<<BSZ, 256, 0, stream>>>(d1o, Wd2, bd2, (float*)d_out);
}
